// APPNP_30279519437686
// MI455X (gfx1250) — matrix operand order repaired and checked by reading, not compiled, not run on hardware
//
#include <hip/hip_runtime.h>
#include <hip/hip_bf16.h>

// ---------------------------------------------------------------------------
// Types for CDNA5 WMMA (wave32)
// ---------------------------------------------------------------------------
typedef __attribute__((ext_vector_type(4)))  float  v4f_t;
typedef __attribute__((ext_vector_type(8)))  float  v8f_t;
typedef __attribute__((ext_vector_type(8)))  __bf16 v8bf_t;
typedef __attribute__((ext_vector_type(16))) __bf16 v16bf_t;

// Global (addrspace 1) load/store helpers: force global_load_* / global_store_*
// instead of flat_* (flat ties up DScnt + the LDS issue path on CDNA5).
#define GLOBAL_AS __attribute__((address_space(1)))
template <typename T>
static __device__ __forceinline__ T gload(const void* p) {
  return *(const GLOBAL_AS T*)(unsigned long long)p;
}
template <typename T>
static __device__ __forceinline__ void gstore(void* p, T v) {
  *(GLOBAL_AS T*)(unsigned long long)p = v;
}

static __device__ __forceinline__ unsigned short f32_to_bf16_bits(float f) {
  unsigned int u = __float_as_uint(f);
  u += 0x7FFFu + ((u >> 16) & 1u);           // round-to-nearest-even
  return (unsigned short)(u >> 16);
}
static __device__ __forceinline__ __bf16 f32_to_bf16(float f) {
  unsigned short b = f32_to_bf16_bits(f);
  __bf16 r;
  __builtin_memcpy(&r, &b, sizeof(r));
  return r;
}

// ---- A-fragment loads: per ISA layout, lane holds row (lane&15) with
//      K = {kb..kb+7} in v0..3 and K = {kb+16..kb+23} in v4..7, kb = 8*(lane>=16)
static __device__ __forceinline__ v16bf_t load_afrag(const __bf16* p) {
  v8bf_t lo = gload<v8bf_t>(p);        // 16B-aligned global_load_b128
  v8bf_t hi = gload<v8bf_t>(p + 16);
  return __builtin_shufflevector(lo, hi, 0, 1, 2, 3, 4, 5, 6, 7,
                                         8, 9, 10, 11, 12, 13, 14, 15);
}
static __device__ __forceinline__ v16bf_t load_afrag(const float* p) {
  v4f_t f0 = gload<v4f_t>(p);          // 4x global_load_b128
  v4f_t f1 = gload<v4f_t>(p + 4);
  v4f_t f2 = gload<v4f_t>(p + 16);
  v4f_t f3 = gload<v4f_t>(p + 20);
  v16bf_t r;
#pragma unroll
  for (int i = 0; i < 4; ++i) {
    r[i]      = f32_to_bf16(f0[i]);
    r[i + 4]  = f32_to_bf16(f1[i]);
    r[i + 8]  = f32_to_bf16(f2[i]);
    r[i + 12] = f32_to_bf16(f3[i]);
  }
  return r;
}

static __device__ __forceinline__ void store_out(float* p, float v) {
  gstore<float>(p, v);
}
static __device__ __forceinline__ void store_out(__bf16* p, float v) {
  gstore<__bf16>(p, f32_to_bf16(v));
}

// ---------------------------------------------------------------------------
// GEMM: out[M,Nc] = act(A[M,K] @ B[K,Nc] + bias), B given pre-transposed as
// Bt[Nc,K] row-major (B fragments use the same two 16B loads per lane as A).
// One wave computes a 16 x (16*NT) output stripe: the A fragment is reused
// across NT back-to-back WMMAs (XDL-friendly), K stepped by 32.
// ---------------------------------------------------------------------------
template <typename AT, typename OT, bool RELU, int NT>
__global__ void gemm_wmma_bf16(const AT* __restrict__ A,
                               const __bf16* __restrict__ Bt,
                               const float* __restrict__ bias,
                               OT* __restrict__ out,
                               int M, int K, int Nc) {
  const int lane = threadIdx.x;                 // 0..31 (wave32)
  const int row0 = blockIdx.x * 16;
  const int col0 = blockIdx.y * (16 * NT);
  const int half = lane >> 4;                   // 0 or 1
  const int arow = row0 + (lane & 15);
  const int ncol = lane & 15;

  const AT* aptr = A + (size_t)arow * K + half * 8;   // A K-split base

  const __bf16* bptr[NT];
#pragma unroll
  for (int j = 0; j < NT; ++j)
    bptr[j] = Bt + (size_t)(col0 + j * 16 + ncol) * K + half * 8;

  const v8f_t vzero = {};
  v8f_t acc[NT];
#pragma unroll
  for (int j = 0; j < NT; ++j) acc[j] = vzero;

  for (int k0 = 0; k0 < K; k0 += 32) {
    __builtin_prefetch(aptr + 64, 0, 0);        // global_prefetch_b8
    v16bf_t a = load_afrag(aptr);
#pragma unroll
    for (int j = 0; j < NT; ++j) {
      v16bf_t b = load_afrag(bptr[j]);          // 2x 16B-aligned global loads, same K split as A
      // v_wmma_f32_16x16x32_bf16: (neg_a, A, neg_b, B, c_mod, C, reuse_a, reuse_b)
      acc[j] = __builtin_amdgcn_wmma_f32_16x16x32_bf16(false, a, false, b,
                                                       (short)0, acc[j],
                                                       false, false);
      bptr[j] += 32;
    }
    aptr += 32;
  }

  // D layout: VGPR r -> row (r + 8*half), column (lane&15)
  const int mrow0 = row0 + half * 8;
#pragma unroll
  for (int j = 0; j < NT; ++j) {
    const int bcol = col0 + j * 16 + ncol;
    const float bv = bias[bcol];
#pragma unroll
    for (int r = 0; r < 8; ++r) {
      float v = acc[j][r] + bv;
      if (RELU) v = fmaxf(v, 0.0f);
      store_out(out + (size_t)(mrow0 + r) * Nc + bcol, v);
    }
  }
}

// ---------------------------------------------------------------------------
// Helper kernels
// ---------------------------------------------------------------------------
__global__ void transpose_f32_to_bf16(const float* __restrict__ in,
                                      __bf16* __restrict__ out, int K, int Nn) {
  int t = blockIdx.x * blockDim.x + threadIdx.x;
  if (t >= K * Nn) return;
  int k = t / Nn;
  int n = t - k * Nn;
  gstore<__bf16>(out + (size_t)n * K + k,
                 f32_to_bf16(gload<float>(in + t)));   // [K,Nn] -> [Nn,K]
}

__global__ void zero_f32(float* __restrict__ p, int n) {
  int t = blockIdx.x * blockDim.x + threadIdx.x;
  if (t < n) gstore<float>(p + t, 0.0f);
}

__global__ void degree_accum(const int* __restrict__ idx,
                             float* __restrict__ deg, int E) {
  int t = blockIdx.x * blockDim.x + threadIdx.x;
  if (t < E) atomicAdd(&deg[gload<int>(idx + t)], 1.0f);
}

__global__ void degree_to_norm(float* __restrict__ d, int N) {
  int t = blockIdx.x * blockDim.x + threadIdx.x;
  if (t < N) {
    float v = gload<float>(d + t);
    gstore<float>(d + t, (v > 0.0f) ? rsqrtf(fmaxf(v, 1.0f)) : 0.0f);
  }
}

// 4 threads per edge (one float4 each): coalesced gather of h[src]*norm_src,
// scatter-add into agg[dst]. C=16.
__global__ void scatter_edges(const float* __restrict__ h,
                              const float* __restrict__ nsrc,
                              const int* __restrict__ src,
                              const int* __restrict__ dst,
                              float* __restrict__ agg, int E) {
  int t = blockIdx.x * blockDim.x + threadIdx.x;
  if (t >= E * 4) return;
  int e = t >> 2;
  int q = t & 3;
  int s = gload<int>(src + e);
  int d = gload<int>(dst + e);
  float ns = gload<float>(nsrc + s);
  v4f_t v = gload<v4f_t>(h + (size_t)s * 16 + q * 4);
  float* ap = agg + (size_t)d * 16 + q * 4;
  atomicAdd(ap + 0, v[0] * ns);
  atomicAdd(ap + 1, v[1] * ns);
  atomicAdd(ap + 2, v[2] * ns);
  atomicAdd(ap + 3, v[3] * ns);
}

// h_out = (1-ALPHA) * agg * norm_dst + ALPHA * h0   (C=16, 4 float4 per node)
__global__ void ppr_update(const float* __restrict__ agg,
                           const float* __restrict__ ndst,
                           const float* __restrict__ h0,
                           float* __restrict__ hout, int N) {
  int t = blockIdx.x * blockDim.x + threadIdx.x;
  if (t >= N * 4) return;
  int n = t >> 2;
  int q = t & 3;
  float nd = gload<float>(ndst + n);
  v4f_t a = gload<v4f_t>(agg + (size_t)n * 16 + q * 4);
  v4f_t z = gload<v4f_t>(h0 + (size_t)n * 16 + q * 4);
  v4f_t r;
#pragma unroll
  for (int i = 0; i < 4; ++i) r[i] = 0.9f * (a[i] * nd) + 0.1f * z[i];
  gstore<v4f_t>(hout + (size_t)n * 16 + q * 4, r);
}

// ---------------------------------------------------------------------------
// Launch
// ---------------------------------------------------------------------------
extern "C" void kernel_launch(void* const* d_in, const int* in_sizes, int n_in,
                              void* d_out, int out_size, void* d_ws,
                              size_t ws_size, hipStream_t stream) {
  (void)n_in; (void)out_size; (void)ws_size;
  const float* features = (const float*)d_in[0];
  const float* W0 = (const float*)d_in[1];
  const float* b0 = (const float*)d_in[2];
  const float* W1 = (const float*)d_in[3];
  const float* b1 = (const float*)d_in[4];
  const float* W2 = (const float*)d_in[5];
  const float* b2 = (const float*)d_in[6];
  const int* src = (const int*)d_in[7];
  const int* dst = (const int*)d_in[8];
  float* out = (float*)d_out;

  const int H   = in_sizes[2];           // 128
  const int C   = in_sizes[6];           // 16
  const int FIN = in_sizes[1] / H;       // 256
  const int N   = in_sizes[0] / FIN;     // 100000 (multiple of 16)
  const int E   = in_sizes[7];           // 1.6M

  // Workspace carve-out (256B aligned)
  char* ws = (char*)d_ws;
  size_t off = 0;
  auto carve = [&](size_t bytes) -> char* {
    char* p = ws + off;
    off = (off + bytes + 255) & ~(size_t)255;
    return p;
  };
  __bf16* W0t  = (__bf16*)carve((size_t)H * FIN * sizeof(__bf16));
  __bf16* W1t  = (__bf16*)carve((size_t)H * H * sizeof(__bf16));
  __bf16* W2t  = (__bf16*)carve((size_t)C * H * sizeof(__bf16));
  __bf16* h1   = (__bf16*)carve((size_t)N * H * sizeof(__bf16));
  __bf16* h2   = (__bf16*)carve((size_t)N * H * sizeof(__bf16));
  float*  h0   = (float*)carve((size_t)N * C * sizeof(float));
  float*  hbuf = (float*)carve((size_t)N * C * sizeof(float));
  float*  agg  = (float*)carve((size_t)N * C * sizeof(float));
  float*  nsrc = (float*)carve((size_t)N * sizeof(float));
  float*  ndst = (float*)carve((size_t)N * sizeof(float));

  const int TB = 256;

  // Weight transpose + bf16 conversion (tiny)
  transpose_f32_to_bf16<<<(FIN * H + TB - 1) / TB, TB, 0, stream>>>(W0, W0t, FIN, H);
  transpose_f32_to_bf16<<<(H * H + TB - 1) / TB, TB, 0, stream>>>(W1, W1t, H, H);
  transpose_f32_to_bf16<<<(H * C + TB - 1) / TB, TB, 0, stream>>>(W2, W2t, H, C);

  // MLP: WMMA bf16-in / f32-accumulate. 16x128 stripe per wave for H-wide
  // outputs (8 WMMAs per A fragment), 16x16 for the C=16 output.
  gemm_wmma_bf16<float, __bf16, true, 8>
      <<<dim3(N / 16, H / 128), 32, 0, stream>>>(features, W0t, b0, h1, N, FIN, H);
  gemm_wmma_bf16<__bf16, __bf16, true, 8>
      <<<dim3(N / 16, H / 128), 32, 0, stream>>>(h1, W1t, b1, h2, N, H, H);
  gemm_wmma_bf16<__bf16, float, false, 1>
      <<<dim3(N / 16, C / 16), 32, 0, stream>>>(h2, W2t, b2, h0, N, H, C);

  // Symmetric normalization
  zero_f32<<<(N + TB - 1) / TB, TB, 0, stream>>>(nsrc, N);
  zero_f32<<<(N + TB - 1) / TB, TB, 0, stream>>>(ndst, N);
  degree_accum<<<(E + TB - 1) / TB, TB, 0, stream>>>(src, nsrc, E);
  degree_accum<<<(E + TB - 1) / TB, TB, 0, stream>>>(dst, ndst, E);
  degree_to_norm<<<(N + TB - 1) / TB, TB, 0, stream>>>(nsrc, N);
  degree_to_norm<<<(N + TB - 1) / TB, TB, 0, stream>>>(ndst, N);

  // K=10 personalized-PageRank propagation
  for (int step = 0; step < 10; ++step) {
    zero_f32<<<(N * C + TB - 1) / TB, TB, 0, stream>>>(agg, N * C);
    const float* hin = (step == 0) ? h0 : hbuf;
    scatter_edges<<<(E * 4 + TB - 1) / TB, TB, 0, stream>>>(hin, nsrc, src, dst, agg, E);
    float* hout = (step == 9) ? out : hbuf;
    ppr_update<<<(N * 4 + TB - 1) / TB, TB, 0, stream>>>(agg, ndst, h0, hout, N);
  }
}
